// GCN_69114613729821
// MI455X (gfx1250) — compile-verified, matrix operand order checked by reading, not run on hardware
//
#include <hip/hip_runtime.h>
#include <hip/hip_bf16.h>

#define NNODES 50000

typedef __attribute__((ext_vector_type(16))) __bf16 v16bf_t;
typedef __attribute__((ext_vector_type(8)))  __bf16 v8bf_t;
typedef __attribute__((ext_vector_type(4)))  __bf16 v4bf_t;
typedef __attribute__((ext_vector_type(8)))  float  v8f_t;
typedef __attribute__((ext_vector_type(4)))  int    v4i_t;
typedef __attribute__((address_space(3)))    v4i_t  lds_v4i_t;

#if defined(__has_builtin)
#if __has_builtin(__builtin_amdgcn_global_load_async_to_lds_b128) && \
    __has_builtin(__builtin_amdgcn_s_wait_asynccnt)
#define USE_ASYNC_LDS 1
#endif
#endif

// ---------------------------------------------------------------- utilities
__global__ void zero_f32_kernel(float* __restrict__ p, int n) {
    int i = blockIdx.x * blockDim.x + threadIdx.x;
    if (i < n) p[i] = 0.0f;
}

// cur[n][0:6] = pos||norm, rest zero-padded (so layer-1 GEMM has K=128)
__global__ void build_x0_kernel(const float* __restrict__ pos,
                                const float* __restrict__ nrm,
                                float* __restrict__ cur) {
    int idx = blockIdx.x * blockDim.x + threadIdx.x;   // over N*128
    if (idx >= NNODES * 128) return;
    int n = idx >> 7, f = idx & 127;
    float v = 0.0f;
    if (f < 3)      v = pos[n * 3 + f];
    else if (f < 6) v = nrm[n * 3 + (f - 3)];
    cur[idx] = v;
}

// Wt[o][k] = bf16(W[k][o]) for k<din else 0  (transposed so B-frags load
// contiguously with the same VGPR packing as A-frags)
__global__ void conv_w_kernel(const float* __restrict__ W, int din,
                              __bf16* __restrict__ Wt) {
    int idx = blockIdx.x * blockDim.x + threadIdx.x;   // over 128*128
    if (idx >= 128 * 128) return;
    int o = idx >> 7, k = idx & 127;
    float v = (k < din) ? W[k * 128 + o] : 0.0f;
    Wt[o * 128 + k] = (__bf16)v;
}

// in-degree counts (shared by all four layers)
__global__ void count_kernel(const int* __restrict__ dst, int E,
                             int* __restrict__ cnt) {
    int e = blockIdx.x * blockDim.x + threadIdx.x;
    if (e < E) atomicAdd(&cnt[dst[e]], 1);
}

// one wave per edge; lane l scatters features [4l,4l+4) via fp32 HW atomics
__global__ void scatter_kernel(const float* __restrict__ x,
                               const int* __restrict__ src,
                               const int* __restrict__ dst,
                               float* __restrict__ agg, int E, int nfeat) {
    int wave = (blockIdx.x * blockDim.x + threadIdx.x) >> 5;
    int lane = threadIdx.x & 31;
    if (wave >= E) return;
    int s = src[wave];
    int d = dst[wave];
    int f = lane * 4;
    if (f < nfeat) {
        const float4 v = *(const float4*)(x + (size_t)s * 128 + f);
        float* p = agg + (size_t)d * 128 + f;
        unsafeAtomicAdd(p + 0, v.x);
        unsafeAtomicAdd(p + 1, v.y);
        unsafeAtomicAdd(p + 2, v.z);
        unsafeAtomicAdd(p + 3, v.w);
    }
}

// xb[n][f] = bf16(agg[n][f] / max(cnt[n],1))
__global__ void scale_cvt_kernel(const float* __restrict__ agg,
                                 const int* __restrict__ cnt,
                                 __bf16* __restrict__ xb) {
    int idx = blockIdx.x * blockDim.x + threadIdx.x;   // over N*32 (4 feats each)
    if (idx >= NNODES * 32) return;
    int n = idx >> 5, c4 = (idx & 31) * 4;
    float inv = 1.0f / fmaxf((float)cnt[n], 1.0f);
    const float4 v = *(const float4*)(agg + (size_t)n * 128 + c4);
    v4bf_t o;
    o[0] = (__bf16)(v.x * inv);
    o[1] = (__bf16)(v.y * inv);
    o[2] = (__bf16)(v.z * inv);
    o[3] = (__bf16)(v.w * inv);
    *(v4bf_t*)(xb + (size_t)n * 128 + c4) = o;
}

// out[32-row tile][128] = relu(xb_tile @ Wt^T + bias), bf16 WMMA, f32 accum.
// Block: 256 threads = 8 waves; wave w owns cols [16w,16w+16) and computes two
// stacked 16x16 M-tiles, reusing each B-fragment for both -> 8 WMMAs/wave.
// A tile (32x128 bf16, 8KB) staged through LDS, via the CDNA5 async-to-LDS
// path when the toolchain exposes it.
__global__ __launch_bounds__(256)
void gemm_wmma_kernel(const __bf16* __restrict__ xb,
                      const __bf16* __restrict__ Wt,
                      const float* __restrict__ bias,
                      float* __restrict__ out) {
    __shared__ __bf16 As[32 * 128];

    const int m0     = blockIdx.x * 32;          // last block is partial
    const int waveId = threadIdx.x >> 5;
    const int lane   = threadIdx.x & 31;

    // stage A tile: each thread moves 32B (two b128 chunks); clamp tail rows
    {
        const int lrow = threadIdx.x >> 3;        // 0..31
        const int lcol = (threadIdx.x & 7) * 16;  // 0..112 (elements)
        int grow = m0 + lrow;
        if (grow > NNODES - 1) grow = NNODES - 1;
        const __bf16* gp = xb + (size_t)grow * 128 + lcol;
        __bf16*       lp = As + lrow * 128 + lcol;
#ifdef USE_ASYNC_LDS
        __builtin_amdgcn_global_load_async_to_lds_b128((v4i_t*)gp,
                                                       (lds_v4i_t*)lp, 0, 0);
        __builtin_amdgcn_global_load_async_to_lds_b128((v4i_t*)(gp + 8),
                                                       (lds_v4i_t*)(lp + 8), 0, 0);
        __builtin_amdgcn_s_wait_asynccnt(0);
#else
        *(v8bf_t*)lp       = *(const v8bf_t*)gp;
        *(v8bf_t*)(lp + 8) = *(const v8bf_t*)(gp + 8);
#endif
    }
    __syncthreads();

    const int n0    = waveId * 16;
    const int nn    = lane & 15;                 // A row (M) / B col (N) index
    const int khalf = lane >> 4;                 // which 8-wide K half this lane packs

    // bias depends only on the output column -> same value in all acc VGPRs
    v8f_t c0, c1;
    const float bv = bias[n0 + nn];
#pragma unroll
    for (int i = 0; i < 8; ++i) { c0[i] = bv; c1[i] = bv; }

#pragma unroll
    for (int kk = 0; kk < 128; kk += 32) {
        // B fragment: Wt row (n0+nn) is column (n0+nn) of W -> same packing as A
        const __bf16* bp = Wt + (size_t)(n0 + nn) * 128 + kk + khalf * 8;
        v8bf_t blo = *(const v8bf_t*)bp;
        v8bf_t bhi = *(const v8bf_t*)(bp + 16);
        // A fragments for the two stacked M-tiles
        v8bf_t a0lo = *(const v8bf_t*)(As + nn * 128 + kk + khalf * 8);
        v8bf_t a0hi = *(const v8bf_t*)(As + nn * 128 + kk + 16 + khalf * 8);
        v8bf_t a1lo = *(const v8bf_t*)(As + (16 + nn) * 128 + kk + khalf * 8);
        v8bf_t a1hi = *(const v8bf_t*)(As + (16 + nn) * 128 + kk + 16 + khalf * 8);

        v16bf_t a0, a1, b;
#pragma unroll
        for (int i = 0; i < 8; ++i) {
            b[i]  = blo[i];  b[i + 8]  = bhi[i];
            a0[i] = a0lo[i]; a0[i + 8] = a0hi[i];
            a1[i] = a1lo[i]; a1[i + 8] = a1hi[i];
        }
        c0 = __builtin_amdgcn_wmma_f32_16x16x32_bf16(
                 false, a0, false, b, (short)0, c0, false, false);
        c1 = __builtin_amdgcn_wmma_f32_16x16x32_bf16(
                 false, a1, false, b, (short)0, c1, false, false);
    }

    // C/D layout: VGPR v, lanes 0-15 -> M=v, lanes 16-31 -> M=v+8
    const int rb = 8 * khalf;
#pragma unroll
    for (int v = 0; v < 8; ++v) {
        const int r0 = m0 + rb + v;          // always < NNODES for this grid
        const int r1 = m0 + 16 + rb + v;     // may overrun in the tail block
        float x0 = c0[v]; x0 = x0 > 0.0f ? x0 : 0.0f;
        float x1 = c1[v]; x1 = x1 > 0.0f ? x1 : 0.0f;
        out[(size_t)r0 * 128 + n0 + nn] = x0;
        if (r1 < NNODES) out[(size_t)r1 * 128 + n0 + nn] = x1;
    }
}

// final [N,128] @ [128,3] + b, lin_w cached in LDS
__global__ void final_linear_kernel(const float* __restrict__ x,
                                    const float* __restrict__ lin_w,
                                    const float* __restrict__ lin_b,
                                    float* __restrict__ out) {
    __shared__ float w[128 * 3];
    __shared__ float bsh[3];
    for (int i = threadIdx.x; i < 384; i += blockDim.x) w[i] = lin_w[i];
    if (threadIdx.x < 3) bsh[threadIdx.x] = lin_b[threadIdx.x];
    __syncthreads();

    int n = blockIdx.x * blockDim.x + threadIdx.x;
    if (n >= NNODES) return;
    float a0 = bsh[0], a1 = bsh[1], a2 = bsh[2];
    const float* xr = x + (size_t)n * 128;
#pragma unroll 4
    for (int k = 0; k < 128; ++k) {
        float xv = xr[k];
        a0 += xv * w[k * 3 + 0];
        a1 += xv * w[k * 3 + 1];
        a2 += xv * w[k * 3 + 2];
    }
    out[n * 3 + 0] = a0;
    out[n * 3 + 1] = a1;
    out[n * 3 + 2] = a2;
}

// ---------------------------------------------------------------- driver
extern "C" void kernel_launch(void* const* d_in, const int* in_sizes, int n_in,
                              void* d_out, int out_size, void* d_ws, size_t ws_size,
                              hipStream_t stream) {
    const float* pos   = (const float*)d_in[0];
    const float* nrm   = (const float*)d_in[1];
    const int*   ei    = (const int*)d_in[2];
    const int    E     = in_sizes[2] / 2;
    const int*   src   = ei;
    const int*   dst   = ei + E;
    const float* W[4]  = {(const float*)d_in[3],  (const float*)d_in[7],
                          (const float*)d_in[11], (const float*)d_in[15]};
    const float* b[4]  = {(const float*)d_in[6],  (const float*)d_in[10],
                          (const float*)d_in[14], (const float*)d_in[18]};
    const float* lin_w = (const float*)d_in[19];
    const float* lin_b = (const float*)d_in[20];
    const int din[4]   = {6, 128, 128, 128};

    // workspace carve-up (all chunks 16B aligned)
    float*  cur = (float*)d_ws;                           // N*128 f32
    float*  agg = cur + (size_t)NNODES * 128;             // N*128 f32
    __bf16* xb  = (__bf16*)(agg + (size_t)NNODES * 128);  // N*128 bf16
    int*    cnt = (int*)(xb + (size_t)NNODES * 128);      // N i32
    __bf16* Wt  = (__bf16*)(cnt + NNODES);                // 4 * 128*128 bf16

    const int NF      = NNODES * 128;
    const int blkNF   = (NF + 255) / 256;
    const int blkCvt  = (NNODES * 32 + 255) / 256;
    const int blkScat = (E * 32 + 255) / 256;             // wave per edge
    const int blkGemm = (NNODES + 31) / 32;               // 32-row tiles

    // in-degree counts (same for every layer)
    zero_f32_kernel<<<(NNODES + 255) / 256, 256, 0, stream>>>((float*)cnt, NNODES);
    count_kernel<<<(E + 255) / 256, 256, 0, stream>>>(dst, E, cnt);

    // padded input features + bf16 transposed weights
    build_x0_kernel<<<blkNF, 256, 0, stream>>>(pos, nrm, cur);
    for (int l = 0; l < 4; ++l)
        conv_w_kernel<<<(128 * 128 + 255) / 256, 256, 0, stream>>>(
            W[l], din[l], Wt + (size_t)l * 128 * 128);

    for (int l = 0; l < 4; ++l) {
        zero_f32_kernel<<<blkNF, 256, 0, stream>>>(agg, NF);
        scatter_kernel<<<blkScat, 256, 0, stream>>>(cur, src, dst, agg, E,
                                                    l == 0 ? 8 : 128);
        scale_cvt_kernel<<<blkCvt, 256, 0, stream>>>(agg, cnt, xb);
        gemm_wmma_kernel<<<blkGemm, 256, 0, stream>>>(
            xb, Wt + (size_t)l * 128 * 128, b[l], cur);
    }

    final_linear_kernel<<<(NNODES + 255) / 256, 256, 0, stream>>>(
        cur, lin_w, lin_b, (float*)d_out);
}
